// MHA_36292473651442
// MI455X (gfx1250) — matrix operand order repaired and checked by reading, not compiled, not run on hardware
//
#include <hip/hip_runtime.h>

typedef __attribute__((ext_vector_type(8)))  float  v8f;
typedef __attribute__((ext_vector_type(16))) __bf16 v16bf;

#define DEV_ __device__ __forceinline__

DEV_ unsigned short f2bf(float f) {
    unsigned int u = __float_as_uint(f);
    u += 0x7FFFu + ((u >> 16) & 1u);            // round-to-nearest-even
    return (unsigned short)(u >> 16);
}

union Frag {
    v16bf v;
    unsigned int u[8];
};

DEV_ v8f wmma_bf16(const Frag& a, const Frag& b, v8f c) {
    return __builtin_amdgcn_wmma_f32_16x16x32_bf16(
        /*neg_a=*/false, a.v, /*neg_b=*/false, b.v,
        /*c_mod=*/(short)0, c, /*reuse_a=*/false, /*reuse_b=*/false);
}

// ---------------------------------------------------------------------------
// GEMM: Y[4096x1024] = A[4096x1024] @ W[1024x1024] + bias
// MODE 0: A fp32, Y scattered bf16 to [B][H][S][64]   (QKV projections)
// MODE 1: A bf16 (attention out), Y fp32 row-major    (output projection)
// Block tile 128x128, BK=32, 8 waves (4x2), 32x64 per wave (2x4 WMMA tiles).
// ---------------------------------------------------------------------------
template<int MODE>
__global__ __launch_bounds__(256) void gemm128x128(
    const float* __restrict__ Af,
    const unsigned short* __restrict__ Ab,
    const float* __restrict__ W,
    const float* __restrict__ bias,
    unsigned short* __restrict__ Yb,
    float* __restrict__ Yf)
{
    __shared__ unsigned short As[128][32];    // [m][k] bf16
    __shared__ unsigned short Bst[128][32];   // [n][k] bf16 (W transposed tile)

    const int tid  = threadIdx.x;
    const int wave = tid >> 5, lane = tid & 31;
    const int l16  = lane & 15, lhi = lane >> 4;
    const int bm = blockIdx.y * 128, bn = blockIdx.x * 128;
    const int wm = (wave >> 1) * 32, wn = (wave & 1) * 64;

    v8f acc[2][4] = {};

    for (int k0 = 0; k0 < 1024; k0 += 32) {
        __syncthreads();
        if (MODE == 0) {
            #pragma unroll
            for (int p = 0; p < 4; ++p) {                 // 128x32 fp32 -> bf16
                int idx = p * 256 + tid;
                int r = idx >> 3, c = (idx & 7) * 4;
                float4 f = *(const float4*)(Af + (size_t)(bm + r) * 1024 + k0 + c);
                As[r][c + 0] = f2bf(f.x); As[r][c + 1] = f2bf(f.y);
                As[r][c + 2] = f2bf(f.z); As[r][c + 3] = f2bf(f.w);
            }
        } else {
            #pragma unroll
            for (int p = 0; p < 8; ++p) {                 // 128x32 bf16 copy
                int idx = p * 256 + tid;
                int r = idx >> 4, c = idx & 15;
                ((unsigned int*)As[r])[c] =
                    ((const unsigned int*)(Ab + (size_t)(bm + r) * 1024 + k0))[c];
            }
        }
        #pragma unroll
        for (int p = 0; p < 4; ++p) {                     // W 32x128 -> Bst[n][k]
            int idx = p * 256 + tid;
            int r = idx >> 5, c = (idx & 31) * 4;
            float4 f = *(const float4*)(W + (size_t)(k0 + r) * 1024 + bn + c);
            Bst[c + 0][r] = f2bf(f.x); Bst[c + 1][r] = f2bf(f.y);
            Bst[c + 2][r] = f2bf(f.z); Bst[c + 3][r] = f2bf(f.w);
        }
        __syncthreads();

        Frag a[2], bf[4];
        #pragma unroll
        for (int mt = 0; mt < 2; ++mt) {
            const unsigned int* src = (const unsigned int*)As[wm + mt * 16 + l16];
            #pragma unroll
            for (int j = 0; j < 8; ++j)
                a[mt].u[j] = src[(j < 4 ? j : j + 4) + lhi * 4];
        }
        #pragma unroll
        for (int nt = 0; nt < 4; ++nt) {
            const unsigned int* src = (const unsigned int*)Bst[wn + nt * 16 + l16];
            #pragma unroll
            for (int j = 0; j < 8; ++j)
                bf[nt].u[j] = src[(j < 4 ? j : j + 4) + lhi * 4];
        }
        #pragma unroll
        for (int mt = 0; mt < 2; ++mt)
            #pragma unroll
            for (int nt = 0; nt < 4; ++nt)
                acc[mt][nt] = wmma_bf16(a[mt], bf[nt], acc[mt][nt]);
    }

    #pragma unroll
    for (int mt = 0; mt < 2; ++mt) {
        #pragma unroll
        for (int nt = 0; nt < 4; ++nt) {
            int n = bn + wn + nt * 16 + l16;
            float bv = bias[n];
            int row0 = bm + wm + mt * 16 + lhi * 8;
            #pragma unroll
            for (int r = 0; r < 8; ++r) {
                float val = acc[mt][nt][r] + bv;
                int row = row0 + r;
                if (MODE == 0) {
                    int b = row >> 11, s = row & 2047;
                    int h = n >> 6,  d = n & 63;
                    Yb[(((size_t)b * 16 + h) * 2048 + s) * 64 + d] = f2bf(val);
                } else {
                    Yf[(size_t)row * 1024 + n] = val;
                }
            }
        }
    }
}

// ---------------------------------------------------------------------------
// Causal flash attention. Grid (S/128, H, B), 256 threads = 8 waves.
// Wave w owns query rows [q0, q0+16). Key blocks of 32, online softmax.
// K tiles are double-buffered in LDS via GLOBAL_LOAD_ASYNC_TO_LDS_B128
// (ASYNCcnt), issued one block ahead of the compute. V is staged manually
// (needs a transpose during staging). Q/K/V in ws as bf16 [B][H][S][64];
// output bf16 [B*S][H*64].
// ---------------------------------------------------------------------------
__global__ __launch_bounds__(256) void attn_flash(
    const unsigned short* __restrict__ Qb,
    const unsigned short* __restrict__ Kb,
    const unsigned short* __restrict__ Vb,
    const unsigned char* __restrict__ mask,
    unsigned short* __restrict__ Ob)
{
    __shared__ unsigned short Kbuf[2][32][64];  // [buf][key][dk], async dest
    __shared__ unsigned short Vt[64][32];       // [dv][key] (transposed)
    __shared__ unsigned short Ps[8][16][32];    // per-wave P bounce

    const int tid  = threadIdx.x;
    const int wave = tid >> 5, lane = tid & 31;
    const int l16  = lane & 15, lhi = lane >> 4;
    const int h = blockIdx.y, b = blockIdx.z;
    const int q0blk = blockIdx.x * 128;
    const int q0 = q0blk + wave * 16;
    const size_t bh = ((size_t)b * 16 + h) * 2048;
    const unsigned char* mk = mask + (size_t)b * 2048;

    const unsigned long long kbase =
        (unsigned long long)(uintptr_t)(Kb + bh * 64);

    // Per-thread piece of one K tile: 4 KB = 256 x b128.
    const int akey = tid >> 4, achunk = (tid & 15) * 8;   // 8 ushorts = 16B

    // Q fragments (dk 0..31 and 32..63)
    Frag qf[2];
    {
        const unsigned int* src = (const unsigned int*)(Qb + (bh + q0 + l16) * 64);
        #pragma unroll
        for (int f = 0; f < 2; ++f)
            #pragma unroll
            for (int j = 0; j < 8; ++j)
                qf[f].u[j] = src[f * 16 + (j < 4 ? j : j + 4) + lhi * 4];
    }

    float m_[8], l_[8];
    v8f acc[4] = {};
    #pragma unroll
    for (int r = 0; r < 8; ++r) { m_[r] = -1e30f; l_[r] = 0.f; }

    const int nblocks = blockIdx.x * 4 + 4;
    const float scale = 0.125f;                 // 1/sqrt(64)

    // Prologue: async-issue K tile 0 into buffer 0.
    {
        unsigned lds  = (unsigned)(uintptr_t)&Kbuf[0][akey][achunk];
        unsigned goff = (unsigned)((akey * 64 + achunk) * 2);
        asm volatile("global_load_async_to_lds_b128 %0, %1, %2"
                     :: "v"(lds), "v"(goff), "s"(kbase) : "memory");
    }

    for (int jb = 0; jb < nblocks; ++jb) {
        const int key0 = jb * 32;
        const int cb = jb & 1;

        __syncthreads();   // compute(jb-1) finished: V and Kbuf[!cb] reusable

        // Stage V tile (transposed) with plain loads.
        #pragma unroll
        for (int p = 0; p < 4; ++p) {
            int idx = p * 256 + tid;
            int kr = idx >> 5, c = idx & 31;
            unsigned int vv = ((const unsigned int*)(Vb + (bh + key0 + kr) * 64))[c];
            Vt[c * 2 + 0][kr] = (unsigned short)(vv & 0xFFFFu);
            Vt[c * 2 + 1][kr] = (unsigned short)(vv >> 16);
        }

        // Our K tile for this block has landed once ASYNCcnt drains.
        asm volatile("s_wait_asynccnt 0" ::: "memory");

        // Overlap: issue next K tile into the other buffer.
        if (jb + 1 < nblocks) {
            unsigned lds  = (unsigned)(uintptr_t)&Kbuf[cb ^ 1][akey][achunk];
            unsigned goff = (unsigned)((((jb + 1) * 32 + akey) * 64 + achunk) * 2);
            asm volatile("global_load_async_to_lds_b128 %0, %1, %2"
                         :: "v"(lds), "v"(goff), "s"(kbase) : "memory");
        }

        __syncthreads();   // all waves' K(jb) awaited, V visible

        if (key0 > q0 + 15) continue;           // fully above causal diagonal

        v8f sc[2];
        #pragma unroll
        for (int nt = 0; nt < 2; ++nt) {        // scores = Q(16x64) @ K^T
            Frag k0f, k1f;
            const unsigned int* src = (const unsigned int*)Kbuf[cb][nt * 16 + l16];
            #pragma unroll
            for (int j = 0; j < 8; ++j) {
                k0f.u[j] = src[(j < 4 ? j : j + 4) + lhi * 4];
                k1f.u[j] = src[16 + (j < 4 ? j : j + 4) + lhi * 4];
            }
            v8f c = {};
            c = wmma_bf16(qf[0], k0f, c);
            c = wmma_bf16(qf[1], k1f, c);
            sc[nt] = c;
        }

        float p0[8], p1[8], tmax[8];
        const int qr0 = q0 + lhi * 8;
        #pragma unroll
        for (int r = 0; r < 8; ++r) {
            int k0i = key0 + l16, k1i = key0 + 16 + l16;
            float s0 = sc[0][r] * scale;
            float s1 = sc[1][r] * scale;
            if (k0i > qr0 + r || mk[k0i]) s0 = -1e30f;
            if (k1i > qr0 + r || mk[k1i]) s1 = -1e30f;
            p0[r] = s0; p1[r] = s1;
            tmax[r] = fmaxf(s0, s1);
        }
        #pragma unroll
        for (int off = 1; off < 16; off <<= 1)
            #pragma unroll
            for (int r = 0; r < 8; ++r)
                tmax[r] = fmaxf(tmax[r], __shfl_xor(tmax[r], off, 16));

        float rs[8];
        #pragma unroll
        for (int r = 0; r < 8; ++r) {
            float mn = fmaxf(fmaxf(m_[r], tmax[r]), -1e28f);
            float al = __expf(m_[r] - mn);
            m_[r] = mn;
            float e0 = __expf(p0[r] - mn);
            float e1 = __expf(p1[r] - mn);
            p0[r] = e0; p1[r] = e1;
            rs[r] = e0 + e1;
            l_[r] *= al;
            #pragma unroll
            for (int t = 0; t < 4; ++t) acc[t][r] *= al;
        }
        #pragma unroll
        for (int off = 1; off < 16; off <<= 1)
            #pragma unroll
            for (int r = 0; r < 8; ++r)
                rs[r] += __shfl_xor(rs[r], off, 16);
        #pragma unroll
        for (int r = 0; r < 8; ++r) l_[r] += rs[r];

        // P (C-layout f32) -> bf16 A-fragment via per-wave LDS bounce
        #pragma unroll
        for (int r = 0; r < 8; ++r) {
            int M = r + lhi * 8;
            Ps[wave][M][l16]      = f2bf(p0[r]);
            Ps[wave][M][16 + l16] = f2bf(p1[r]);
        }
        asm volatile("s_wait_dscnt 0" ::: "memory");
        Frag pf;
        {
            const unsigned int* src = (const unsigned int*)Ps[wave][l16];
            #pragma unroll
            for (int j = 0; j < 8; ++j)
                pf.u[j] = src[(j < 4 ? j : j + 4) + lhi * 4];
        }
        #pragma unroll
        for (int t = 0; t < 4; ++t) {           // O += P(16x32) @ V(32x64)
            Frag vf;
            const unsigned int* src = (const unsigned int*)Vt[t * 16 + l16];
            #pragma unroll
            for (int j = 0; j < 8; ++j) vf.u[j] = src[(j < 4 ? j : j + 4) + lhi * 4];
            acc[t] = wmma_bf16(pf, vf, acc[t]);
        }
    }

    float inv[8];
    #pragma unroll
    for (int r = 0; r < 8; ++r) inv[r] = (l_[r] > 0.f) ? 1.f / l_[r] : 0.f;
    #pragma unroll
    for (int t = 0; t < 4; ++t)
        #pragma unroll
        for (int r = 0; r < 8; ++r) {
            int row = b * 2048 + q0 + r + lhi * 8;
            int col = h * 64 + t * 16 + l16;
            Ob[(size_t)row * 1024 + col] = f2bf(acc[t][r] * inv[r]);
        }
}

// ---------------------------------------------------------------------------
extern "C" void kernel_launch(void* const* d_in, const int* in_sizes, int n_in,
                              void* d_out, int out_size, void* d_ws, size_t ws_size,
                              hipStream_t stream)
{
    const float* q  = (const float*)d_in[0];
    const float* k  = (const float*)d_in[1];
    const float* v  = (const float*)d_in[2];
    const unsigned char* mask = (const unsigned char*)d_in[3];
    const float* Wq = (const float*)d_in[4];
    const float* bq = (const float*)d_in[5];
    const float* Wk = (const float*)d_in[6];
    const float* bk = (const float*)d_in[7];
    const float* Wv = (const float*)d_in[8];
    const float* bv = (const float*)d_in[9];
    const float* Wo = (const float*)d_in[10];
    const float* bo = (const float*)d_in[11];
    float* out = (float*)d_out;

    const size_t QKV_ELEMS = (size_t)2 * 16 * 2048 * 64;   // 4M bf16 each
    unsigned short* Qw = (unsigned short*)d_ws;
    unsigned short* Kw = Qw + QKV_ELEMS;
    unsigned short* Vw = Kw + QKV_ELEMS;
    unsigned short* Ow = Vw + QKV_ELEMS;                   // [4096][1024] bf16

    dim3 gb(256, 1, 1);
    dim3 gg(8, 32, 1);                                     // N/128 x M/128
    gemm128x128<0><<<gg, gb, 0, stream>>>(q, nullptr, Wq, bq, Qw, nullptr);
    gemm128x128<0><<<gg, gb, 0, stream>>>(k, nullptr, Wk, bk, Kw, nullptr);
    gemm128x128<0><<<gg, gb, 0, stream>>>(v, nullptr, Wv, bv, Vw, nullptr);

    dim3 ga(16, 16, 2);                                    // S/128 x H x B
    attn_flash<<<ga, gb, 0, stream>>>(Qw, Kw, Vw, mask, Ow);

    gemm128x128<1><<<gg, gb, 0, stream>>>(nullptr, Ow, Wo, bo, nullptr, out);
}
